// DecoderRNN_10342281248841
// MI455X (gfx1250) — matrix operand order repaired and checked by reading, not compiled, not run on hardware
//
#include <hip/hip_runtime.h>
#include <hip/hip_bf16.h>
#include <math.h>

// ---------------- problem constants ----------------
#define T_STEPS 128
#define BATCH   256
#define DIN     2048
#define E_DIM   100
#define H_DIM   1024
#define C_DIM   151
#define SIXH    6144
#define FIVEH   5120
#define D_FULL  2148            // DIN + E
#define KX      2048            // K of the big seq GEMM
#define KC_X    (KX / 32)       // 64 K-chunks
#define KCOMB   1152            // K of per-step GEMM (1024 state + 100 emb + 28 pad)
#define KC_C    (KCOMB / 32)    // 36 K-chunks
#define NT      (SIXH / 16)     // 384 N-tiles of 16
#define MT_SEQ  (T_STEPS * BATCH / 16)   // 2048 M-tiles of the big GEMM

typedef __attribute__((ext_vector_type(16))) __bf16 v16bf;
typedef __attribute__((ext_vector_type(8)))  float  v8f;

// ---------------------------------------------------------------------------
// Packed-fragment index helpers (exact CDNA5 WMMA VGPR layouts).
//
// A 16x32 bf16 fragment: lane = (m&15) | (((k>>3)&1)<<4),
//                        elem = ((k>>4)<<3) | (k&7)      (k local to 32-chunk)
// B 32x16 bf16 fragment: lane = (n&15) | (((k>>3)&1)<<4),
//                        elem = ((k>>4)<<3) | (k&7)
// One fragment = 32 lanes x 16 bf16 = 512 elements, stored contiguously so a
// wave reads its whole fragment with a single aligned 32-byte load per lane.
// ---------------------------------------------------------------------------
__device__ __forceinline__ size_t apack_idx(int m, int k, int KC) {
    int mt = m >> 4, kc = k >> 5, kl = k & 31;
    int lane = (m & 15) | (((kl >> 3) & 1) << 4);
    int e    = ((kl >> 4) << 3) | (kl & 7);
    return ((size_t)(mt * KC + kc) * 32 + lane) * 16 + e;
}

// ---------------------------------------------------------------------------
// Kernel 1: pack seq (fp32 row-major [32768 x 2048]) -> bf16 A-fragment layout
// ---------------------------------------------------------------------------
__global__ void pack_a_seq(const float* __restrict__ seq, __bf16* __restrict__ Ap) {
    size_t i = (size_t)blockIdx.x * blockDim.x + threadIdx.x;
    if (i >= (size_t)MT_SEQ * KC_X * 32 * 16) return;
    int e    = (int)(i & 15);
    int lane = (int)((i >> 4) & 31);
    int kc   = (int)((i >> 9) & 63);          // KC_X == 64
    int mt   = (int)(i >> 15);
    int m = (mt << 4) + (lane & 15);
    int k = (kc << 5) + ((e >> 3) << 4) + ((lane >> 4) << 3) + (e & 7);
    Ap[i] = (__bf16)seq[(size_t)m * KX + k];
}

// ---------------------------------------------------------------------------
// Kernel 2: pack W_in[:, :2048] -> bf16 B-fragment layout  (B[k,n] = W_in[n,k])
// ---------------------------------------------------------------------------
__global__ void pack_bx(const float* __restrict__ W_in, __bf16* __restrict__ Bx) {
    size_t i = (size_t)blockIdx.x * blockDim.x + threadIdx.x;
    if (i >= (size_t)NT * KC_X * 32 * 16) return;
    int e    = (int)(i & 15);
    int lane = (int)((i >> 4) & 31);
    int kc   = (int)((i >> 9) & 63);
    int jt   = (int)(i >> 15);
    int k = (kc << 5) + ((e >> 3) << 4) + ((lane >> 4) << 3) + (e & 7);
    int n = (jt << 4) + (lane & 15);
    Bx[i] = (__bf16)W_in[(size_t)n * D_FULL + k];
}

// ---------------------------------------------------------------------------
// Kernel 3: pack combined recurrent B (K=1152, N=6144):
//   k    0..1023 : W_state^T (zero for n >= 5120)
//   k 1024..1123 : W_in[:, 2048:2148]^T (embedding columns)
//   k 1124..1151 : zero pad
// ---------------------------------------------------------------------------
__global__ void pack_bc(const float* __restrict__ W_state,
                        const float* __restrict__ W_in,
                        __bf16* __restrict__ Bc) {
    size_t i = (size_t)blockIdx.x * blockDim.x + threadIdx.x;
    if (i >= (size_t)NT * KC_C * 32 * 16) return;
    int e    = (int)(i & 15);
    int lane = (int)((i >> 4) & 31);
    int jkc  = (int)(i >> 9);
    int kc   = jkc % KC_C;
    int jt   = jkc / KC_C;
    int k = (kc << 5) + ((e >> 3) << 4) + ((lane >> 4) << 3) + (e & 7);
    int n = (jt << 4) + (lane & 15);
    float v = 0.f;
    if (k < H_DIM) {
        if (n < FIVEH) v = W_state[(size_t)n * H_DIM + k];
    } else if (k < H_DIM + E_DIM) {
        v = W_in[(size_t)n * D_FULL + DIN + (k - H_DIM)];
    }
    Bc[i] = (__bf16)v;
}

// ---------------------------------------------------------------------------
// Kernel 4: init carry. Packed A for step 0: state0=0, emb0=obj_embed[0],
// pad=0. Also zero the cell state.
// ---------------------------------------------------------------------------
__global__ void init_state(__bf16* __restrict__ Ap, float* __restrict__ Mem,
                           const float* __restrict__ obj_embed) {
    int i = blockIdx.x * blockDim.x + threadIdx.x;
    if (i < BATCH / 16 * KC_C * 32 * 16) {
        int e    = i & 15;
        int lane = (i >> 4) & 31;
        int jkc  = i >> 9;
        int kc   = jkc % KC_C;
        int k = (kc << 5) + ((e >> 3) << 4) + ((lane >> 4) << 3) + (e & 7);
        float v = 0.f;
        if (k >= H_DIM && k < H_DIM + E_DIM) v = obj_embed[k - H_DIM];
        Ap[i] = (__bf16)v;
    }
    if (i < BATCH * H_DIM) Mem[i] = 0.f;
}

// ---------------------------------------------------------------------------
// Kernel 5: big parallel GEMM  Pix[32768,6144] = seq x W_in[:, :2048]^T.
// 256 threads = 8 waves (2M x 4N); wave tile 32x64; all operands pre-packed
// bf16 -> inner loop is pure b128 loads + 8x v_wmma_f32_16x16x32_bf16.
// ---------------------------------------------------------------------------
__global__ __launch_bounds__(256)
void gemm_seq_win(const v16bf* __restrict__ Ap, const v16bf* __restrict__ Bp,
                  __bf16* __restrict__ Pix) {
    int lane = threadIdx.x & 31;
    int wave = threadIdx.x >> 5;
    int m0 = blockIdx.x * 64 + (wave >> 2) * 32;
    int n0 = blockIdx.y * 256 + (wave & 3) * 64;
    int mt0 = m0 >> 4;
    int jt0 = n0 >> 4;

    v8f acc[2][4];
#pragma unroll
    for (int mi = 0; mi < 2; ++mi)
#pragma unroll
        for (int ni = 0; ni < 4; ++ni) acc[mi][ni] = (v8f){0.f,0.f,0.f,0.f,0.f,0.f,0.f,0.f};

    for (int kc = 0; kc < KC_X; ++kc) {
        v16bf a0 = Ap[((size_t)(mt0 + 0) * KC_X + kc) * 32 + lane];
        v16bf a1 = Ap[((size_t)(mt0 + 1) * KC_X + kc) * 32 + lane];
        v16bf b[4];
#pragma unroll
        for (int ni = 0; ni < 4; ++ni)
            b[ni] = Bp[((size_t)(jt0 + ni) * KC_X + kc) * 32 + lane];
        if (kc + 2 < KC_X) {
            __builtin_prefetch(&Ap[((size_t)mt0 * KC_X + kc + 2) * 32 + lane], 0, 1);
            __builtin_prefetch(&Bp[((size_t)jt0 * KC_X + kc + 2) * 32 + lane], 0, 1);
        }
#pragma unroll
        for (int ni = 0; ni < 4; ++ni) {
            acc[0][ni] = __builtin_amdgcn_wmma_f32_16x16x32_bf16(
                false, a0, false, b[ni], (short)0, acc[0][ni], false, false);
            acc[1][ni] = __builtin_amdgcn_wmma_f32_16x16x32_bf16(
                false, a1, false, b[ni], (short)0, acc[1][ni], false, false);
        }
    }
    // C/D layout: lane holds col n0+(lane&15), rows (lane>>4)*8 + r
    int rbase = (lane >> 4) * 8;
#pragma unroll
    for (int mi = 0; mi < 2; ++mi)
#pragma unroll
        for (int ni = 0; ni < 4; ++ni) {
            int col = n0 + ni * 16 + (lane & 15);
#pragma unroll
            for (int r = 0; r < 8; ++r) {
                int row = m0 + mi * 16 + rbase + r;
                Pix[(size_t)row * SIXH + col] = (__bf16)acc[mi][ni][r];
            }
        }
}

// ---------------------------------------------------------------------------
// Kernel 6: per-step recurrent GEMM.
// G[b,j] = [state|emb] x Bc + Pix[t*B+b, j] + b_in[j] + (j<5120 ? b_state[j] : 0)
// ---------------------------------------------------------------------------
__global__ __launch_bounds__(256)
void gemm_step(const v16bf* __restrict__ Ap, const v16bf* __restrict__ Bp,
               const __bf16* __restrict__ Pix, const float* __restrict__ b_in,
               const float* __restrict__ b_state, float* __restrict__ G, int t) {
    int lane = threadIdx.x & 31;
    int wave = threadIdx.x >> 5;
    int m0 = blockIdx.x * 64 + (wave >> 2) * 32;
    int n0 = blockIdx.y * 256 + (wave & 3) * 64;
    int mt0 = m0 >> 4;
    int jt0 = n0 >> 4;

    v8f acc[2][4];
#pragma unroll
    for (int mi = 0; mi < 2; ++mi)
#pragma unroll
        for (int ni = 0; ni < 4; ++ni) acc[mi][ni] = (v8f){0.f,0.f,0.f,0.f,0.f,0.f,0.f,0.f};

    for (int kc = 0; kc < KC_C; ++kc) {
        v16bf a0 = Ap[((size_t)(mt0 + 0) * KC_C + kc) * 32 + lane];
        v16bf a1 = Ap[((size_t)(mt0 + 1) * KC_C + kc) * 32 + lane];
        v16bf b[4];
#pragma unroll
        for (int ni = 0; ni < 4; ++ni)
            b[ni] = Bp[((size_t)(jt0 + ni) * KC_C + kc) * 32 + lane];
#pragma unroll
        for (int ni = 0; ni < 4; ++ni) {
            acc[0][ni] = __builtin_amdgcn_wmma_f32_16x16x32_bf16(
                false, a0, false, b[ni], (short)0, acc[0][ni], false, false);
            acc[1][ni] = __builtin_amdgcn_wmma_f32_16x16x32_bf16(
                false, a1, false, b[ni], (short)0, acc[1][ni], false, false);
        }
    }
    int rbase = (lane >> 4) * 8;
#pragma unroll
    for (int mi = 0; mi < 2; ++mi)
#pragma unroll
        for (int ni = 0; ni < 4; ++ni) {
            int col = n0 + ni * 16 + (lane & 15);
            float bias = b_in[col] + (col < FIVEH ? b_state[col] : 0.f);
#pragma unroll
            for (int r = 0; r < 8; ++r) {
                int row = m0 + mi * 16 + rbase + r;
                float v = acc[mi][ni][r] + bias
                        + (float)Pix[(size_t)(t * BATCH + row) * SIXH + col];
                G[(size_t)row * SIXH + col] = v;
            }
        }
}

// ---------------------------------------------------------------------------
// Kernel 7: LSTM gates, one thread per (b,h). Updates cell in place; writes
// the fp32 hidden (for the class head) and scatters the bf16 hidden directly
// into the next step's packed A fragments.
// ---------------------------------------------------------------------------
__device__ __forceinline__ float sigmoidf_(float x) { return 1.f / (1.f + __expf(-x)); }

__global__ __launch_bounds__(256)
void gates_kernel(const float* __restrict__ G, float* __restrict__ Mem,
                  float* __restrict__ Hout, __bf16* __restrict__ Ap) {
    int i = blockIdx.x * blockDim.x + threadIdx.x;   // 0 .. 256*1024-1
    int b = i >> 10, h = i & (H_DIM - 1);
    const float* g = G + (size_t)b * SIXH;
    float ig = sigmoidf_(g[h]);
    float fg = sigmoidf_(g[H_DIM + h]);
    float mi = tanhf(g[2 * H_DIM + h]);
    float og = sigmoidf_(g[3 * H_DIM + h]);
    float mem = ig * mi + fg * Mem[i];
    float outv = og * tanhf(mem);
    float hg = sigmoidf_(g[4 * H_DIM + h]);
    outv = hg * outv + (1.f - hg) * g[5 * H_DIM + h];
    Mem[i] = mem;
    Hout[i] = outv;
    Ap[apack_idx(b, h, KC_C)] = (__bf16)outv;
}

// ---------------------------------------------------------------------------
// Kernel 8: class head pred = hid @ W_out^T + b_out (C=151), argmax over
// classes 1..150, label commitment, embedding gather scattered into packed A.
// One block per batch row.
// ---------------------------------------------------------------------------
__global__ __launch_bounds__(256)
void pred_kernel(const float* __restrict__ Hout, const float* __restrict__ W_out,
                 const float* __restrict__ b_out, const int* __restrict__ labels,
                 const float* __restrict__ obj_embed, float* __restrict__ dists,
                 int* __restrict__ comms, __bf16* __restrict__ Ap, int t) {
    __shared__ float hid[H_DIM];
    __shared__ float sval[256];
    __shared__ int   sidx[256];
    __shared__ int   comm_s;
    int b = blockIdx.x, tid = threadIdx.x;
    const float* arow = Hout + (size_t)b * H_DIM;
    for (int i = tid; i < H_DIM; i += 256) hid[i] = arow[i];
    __syncthreads();

    float p = -3.4e38f;
    if (tid < C_DIM) {
        const float* w = W_out + (size_t)tid * H_DIM;
        float s = b_out[tid];
#pragma unroll 4
        for (int k = 0; k < H_DIM; ++k) s += hid[k] * w[k];
        p = s;
        dists[(size_t)(t * BATCH + b) * C_DIM + tid] = s;
    }
    sval[tid] = (tid >= 1 && tid < C_DIM) ? p : -3.4e38f;
    sidx[tid] = tid;
    __syncthreads();
    for (int s = 128; s > 0; s >>= 1) {      // strict '>' keeps lowest index on ties
        if (tid < s && sval[tid + s] > sval[tid]) {
            sval[tid] = sval[tid + s];
            sidx[tid] = sidx[tid + s];
        }
        __syncthreads();
    }
    if (tid == 0) {
        int lab = labels[t * BATCH + b];
        int c = (lab == 0) ? sidx[0] : lab;
        comm_s = c;
        comms[t * BATCH + b] = c;
    }
    __syncthreads();
    if (tid < E_DIM)
        Ap[apack_idx(b, H_DIM + tid, KC_C)] =
            (__bf16)obj_embed[(size_t)(comm_s + 1) * E_DIM + tid];
}

// ---------------------------------------------------------------------------
// host launcher
// ---------------------------------------------------------------------------
extern "C" void kernel_launch(void* const* d_in, const int* in_sizes, int n_in,
                              void* d_out, int out_size, void* d_ws, size_t ws_size,
                              hipStream_t stream) {
    const float* seq       = (const float*)d_in[0];
    const int*   labels    = (const int*)  d_in[1];
    const float* W_in      = (const float*)d_in[2];
    const float* b_in      = (const float*)d_in[3];
    const float* W_state   = (const float*)d_in[4];
    const float* b_state   = (const float*)d_in[5];
    const float* W_out     = (const float*)d_in[6];
    const float* b_out     = (const float*)d_in[7];
    const float* obj_embed = (const float*)d_in[8];

    float* dists = (float*)d_out;
    int*   comms = (int*)((float*)d_out + (size_t)T_STEPS * BATCH * C_DIM);

    // workspace carve-up (all region sizes are multiples of 256B)
    char* ws = (char*)d_ws;
    __bf16* Bx    = (__bf16*)ws;  ws += (size_t)NT * KC_X * 512 * sizeof(__bf16);        // 25.2 MB
    __bf16* Bc    = (__bf16*)ws;  ws += (size_t)NT * KC_C * 512 * sizeof(__bf16);        // 14.2 MB
    __bf16* ApSeq = (__bf16*)ws;  ws += (size_t)MT_SEQ * KC_X * 512 * sizeof(__bf16);    // 134 MB
    __bf16* ApStp = (__bf16*)ws;  ws += (size_t)(BATCH / 16) * KC_C * 512 * sizeof(__bf16);
    __bf16* Pix   = (__bf16*)ws;  ws += (size_t)T_STEPS * BATCH * SIXH * sizeof(__bf16); // 403 MB
    float*  Hout  = (float*)ws;   ws += (size_t)BATCH * H_DIM * sizeof(float);
    float*  G     = (float*)ws;   ws += (size_t)BATCH * SIXH * sizeof(float);
    float*  Mem   = (float*)ws;   ws += (size_t)BATCH * H_DIM * sizeof(float);

    // --- one-time prep (runs every call; deterministic) ---
    {
        size_t n = (size_t)MT_SEQ * KC_X * 32 * 16;
        pack_a_seq<<<dim3((unsigned)((n + 255) / 256)), dim3(256), 0, stream>>>(seq, ApSeq);
    }
    {
        size_t n = (size_t)NT * KC_X * 32 * 16;
        pack_bx<<<dim3((unsigned)((n + 255) / 256)), dim3(256), 0, stream>>>(W_in, Bx);
    }
    {
        size_t n = (size_t)NT * KC_C * 32 * 16;
        pack_bc<<<dim3((unsigned)((n + 255) / 256)), dim3(256), 0, stream>>>(W_state, W_in, Bc);
    }
    init_state<<<dim3((BATCH / 16 * KC_C * 512 + 255) / 256), dim3(256), 0, stream>>>(ApStp, Mem, obj_embed);

    // --- big parallel GEMM over all timesteps ---
    gemm_seq_win<<<dim3(T_STEPS * BATCH / 64, SIXH / 256), dim3(256), 0, stream>>>(
        (const v16bf*)ApSeq, (const v16bf*)Bx, Pix);

    // --- 128 sequential recurrent steps ---
    for (int t = 0; t < T_STEPS; ++t) {
        gemm_step<<<dim3(BATCH / 64, SIXH / 256), dim3(256), 0, stream>>>(
            (const v16bf*)ApStp, (const v16bf*)Bc, Pix, b_in, b_state, G, t);
        gates_kernel<<<dim3(BATCH * H_DIM / 256), dim3(256), 0, stream>>>(G, Mem, Hout, ApStp);
        pred_kernel<<<dim3(BATCH), dim3(256), 0, stream>>>(
            Hout, W_out, b_out, labels, obj_embed, dists, comms, ApStp, t);
    }
}
